// GunnarODE_70085276336946
// MI455X (gfx1250) — matrix operand order repaired and checked by reading, not compiled, not run on hardware
//
#include <hip/hip_runtime.h>
#include <hip/hip_bf16.h>
#include <math.h>

// L=256, B=256, H=256, Y=32, C=16
// Scan is independent per batch row -> 16 workgroups of 16 rows each, no sync.

typedef __attribute__((ext_vector_type(16))) _Float16 v16h;
typedef __attribute__((ext_vector_type(8)))  _Float16 v8h;
typedef __attribute__((ext_vector_type(8)))  float    v8f;

__device__ __forceinline__ v8f wmma_f32(v16h a, v16h b, v8f c) {
  return __builtin_amdgcn_wmma_f32_16x16x32_f16(
      /*neg_a=*/false, a, /*neg_b=*/false, b,
      /*c_mod=*/(short)0, c, /*reuse_a=*/false, /*reuse_b=*/false);
}

// A-fragment (16x32 f16, M x K) from an LDS row-major f16 matrix [16][ldk].
// lane l: M = l&15, hi = l>>4; j=0..15 -> K=k0+hi*16+j
__device__ __forceinline__ v16h load_afrag_lds(const _Float16* base, int ldk, int k0, int lane) {
  int M  = lane & 15;
  int hi = lane >> 4;
  const _Float16* p = base + M * ldk + k0 + hi * 16;
  union { v16h v; v8h h[2]; } u;
  u.h[0] = *(const v8h*)p;
  u.h[1] = *(const v8h*)(p + 8);
  return u.v;
}

// B-fragment (32x16 f16, K x N) from K-contiguous (pre-transposed) f16 Wt[n][k].
// lane l: N = n0+(l&15), K = k0+(l>>4)*16 + (0..15)  -> 16 contiguous halves
__device__ __forceinline__ v16h load_bfrag_g(const _Float16* __restrict__ Wt,
                                             int n0, int k0, int ldk, int lane) {
  int n  = n0 + (lane & 15);
  int kb = k0 + ((lane >> 4) << 4);
  const _Float16* p = Wt + (size_t)n * ldk + kb;
  union { v16h v; v8h h[2]; } u;
  u.h[0] = *(const v8h*)p;
  u.h[1] = *(const v8h*)(p + 8);
  return u.v;
}

__device__ __forceinline__ v16h load_bfrag_lds(const _Float16* Wl,
                                               int n0, int k0, int ldk, int lane) {
  int n  = n0 + (lane & 15);
  int kb = k0 + ((lane >> 4) << 4);
  const _Float16* p = Wl + n * ldk + kb;
  union { v16h v; v8h h[2]; } u;
  u.h[0] = *(const v8h*)p;
  u.h[1] = *(const v8h*)(p + 8);
  return u.v;
}

// Async copy of one 16B chunk: global -> LDS (dynamic LDS starts at LDS offset 0).
__device__ __forceinline__ void async_b128_to_lds(unsigned lds_off, const void* gptr) {
  unsigned long long ga = (unsigned long long)(size_t)gptr;
  asm volatile("global_load_async_to_lds_b128 %0, %1, off"
               :: "v"(lds_off), "v"(ga) : "memory");
}
__device__ __forceinline__ void wait_asynccnt0() {
  asm volatile("s_wait_asynccnt 0" ::: "memory");
}

// ---------- prep kernels ----------

// dst[n*K + k] = (f16) src[k*N + n]   (src is row-major K x N)
__global__ void transpose_cast_kernel(const float* __restrict__ src,
                                      _Float16* __restrict__ dst, int K, int N) {
  int idx = blockIdx.x * blockDim.x + threadIdx.x;
  if (idx >= K * N) return;
  int n = idx / K;
  int k = idx - n * K;
  dst[idx] = (_Float16)src[k * N + n];
}

// D[s][b][c]: derivative vector per scan step (tau==0 collapses Hermite to dnode).
__global__ void make_dnode_kernel(const float* __restrict__ us, float* __restrict__ D) {
  int idx = blockIdx.x * blockDim.x + threadIdx.x;
  if (idx >= 255 * 256 * 16) return;
  int c = idx & 15;
  int b = (idx >> 4) & 255;
  int s = idx >> 12;
  float v;
  if (c == 0) {
    v = 1.0f;   // time channel diff
  } else {
    int l1 = (s == 0) ? 1 : s;
    int l0 = l1 - 1;
    v = us[((size_t)l1 * 256 + b) * 15 + (c - 1)] -
        us[((size_t)l0 * 256 + b) * 15 + (c - 1)];
  }
  D[idx] = v;
}

// ---------- scan kernel: 16 blocks x 512 threads (16 waves), one block per 16 batch rows ----------
// Dynamic LDS layout (177 KB, one block per WGP):
static const unsigned OFF_W1   = 0;        // 256*256 f16 = 131072
static const unsigned OFF_ZF   = 131072;   // 16*256 f32  = 16384
static const unsigned OFF_ZH   = 147456;   // 16*256 f16  = 8192
static const unsigned OFF_HID  = 155648;   // 16*256 f16  = 8192
static const unsigned OFF_DZ   = 163840;   // 16*256 f32  = 16384
static const unsigned OFF_DLOC = 180224;   // 16*16  f32  = 1024
static const unsigned SCAN_SMEM = 181248;

__global__ __launch_bounds__(512) void scan_kernel(
    const float* __restrict__ h0, const _Float16* __restrict__ W1t,
    const float* __restrict__ b1, const _Float16* __restrict__ W2t,
    const float* __restrict__ b2, const float* __restrict__ Dmat,
    float* __restrict__ zseq)
{
  extern __shared__ char smem[];
  _Float16* W1l  = (_Float16*)(smem + OFF_W1);
  float*    zf   = (float*)   (smem + OFF_ZF);
  _Float16* zh   = (_Float16*)(smem + OFF_ZH);
  _Float16* hidh = (_Float16*)(smem + OFF_HID);
  float*    dz   = (float*)   (smem + OFF_DZ);
  float*    dloc = (float*)   (smem + OFF_DLOC);

  const int tid  = threadIdx.x;
  const int lane = tid & 31;
  const int wv   = tid >> 5;            // 0..15
  const int b0   = blockIdx.x * 16;     // batch row block

  // Stage W1t (128 KB f16) into LDS once via async-to-LDS: 512 threads x 16 B128.
  #pragma unroll
  for (int i = 0; i < 16; ++i) {
    unsigned off = (unsigned)((i * 512 + tid) * 16);
    async_b128_to_lds(OFF_W1 + off, (const char*)W1t + off);
  }

  // init z = h0 rows; emit z_seq[0]
  {
    int row = tid >> 5, c0 = (tid & 31) * 8;
    for (int i = 0; i < 8; ++i) {
      float v = h0[(size_t)(b0 + row) * 256 + c0 + i];
      zf[row * 256 + c0 + i] = v;
      zh[row * 256 + c0 + i] = (_Float16)v;
      zseq[(size_t)(b0 + row) * 256 + c0 + i] = v;   // l = 0
    }
  }
  wait_asynccnt0();
  __syncthreads();

  for (int s = 0; s < 255; ++s) {
    if (tid < 256)
      dloc[tid] = Dmat[(size_t)s * 4096 + (size_t)(b0 + (tid >> 4)) * 16 + (tid & 15)];
    __syncthreads();

    v16h afr[8];

    // ---- GEMM1: hid = relu(z @ W1 + b1), W1 B-fragments from LDS ----
    #pragma unroll
    for (int kk = 0; kk < 8; ++kk) afr[kk] = load_afrag_lds(zh, 256, kk * 32, lane);
    {
      int n0 = wv * 16;                 // 16 waves x 1 tile = N=256
      v8f acc = {};
      #pragma unroll
      for (int kk = 0; kk < 8; ++kk) {
        v16h bf = load_bfrag_lds(W1l, n0, kk * 32, 256, lane);
        acc = wmma_f32(afr[kk], bf, acc);
      }
      int n = n0 + (lane & 15);
      float bb = b1[n];
      #pragma unroll
      for (int r = 0; r < 8; ++r) {
        int M = r + (lane >> 4) * 8;
        float v = acc[r] + bb;
        hidh[M * 256 + n] = (_Float16)(v > 0.f ? v : 0.f);
      }
    }
    __syncthreads();

    // ---- GEMM2: out = tanh(hid @ W2 + b2); dz[b,h] = sum_c out[b,h*16+c] * d[b,c] ----
    #pragma unroll
    for (int kk = 0; kk < 8; ++kk) afr[kk] = load_afrag_lds(hidh, 256, kk * 32, lane);
    for (int t = 0; t < 16; ++t) {
      int h  = wv * 16 + t;             // each 16-wide N tile == one h (C == 16)
      int n0 = h * 16;
      if (t + 1 < 16)
        __builtin_prefetch(W2t + (size_t)(n0 + 16) * 256, 0, 1);
      v8f acc = {};
      #pragma unroll
      for (int kk = 0; kk < 8; ++kk) {
        v16h bf = load_bfrag_g(W2t, n0, kk * 32, 256, lane);
        acc = wmma_f32(afr[kk], bf, acc);
      }
      int c = lane & 15;
      float bb = b2[n0 + c];
      #pragma unroll
      for (int r = 0; r < 8; ++r) {
        int M = r + (lane >> 4) * 8;
        float v = tanhf(acc[r] + bb) * dloc[M * 16 + c];
        // reduce over c == the 16 lanes of each half-wave (halves carry distinct M)
        v += __shfl_xor(v, 1, 16);
        v += __shfl_xor(v, 2, 16);
        v += __shfl_xor(v, 4, 16);
        v += __shfl_xor(v, 8, 16);
        if (c == 0) dz[M * 256 + h] = v;   // each (M,h) owned by exactly one wave
      }
    }
    __syncthreads();

    // ---- z update (dt == 1) + store z_seq[s+1] ----
    {
      int row = tid >> 5, c0 = (tid & 31) * 8;
      size_t orow = ((size_t)(s + 1) * 256 + b0 + row) * 256;
      for (int i = 0; i < 8; ++i) {
        int idx = row * 256 + c0 + i;
        float nz = zf[idx] + dz[idx];
        zf[idx] = nz;
        zh[idx] = (_Float16)nz;
        zseq[orow + c0 + i] = nz;
      }
    }
    __syncthreads();
  }
}

// ---------- readout: out = tanh(z_seq @ dW1 + db1) @ dW2 + db2 ----------
// 65536 rows -> 4096 blocks x 16 rows, 256 threads (8 waves) each.

__global__ __launch_bounds__(256) void readout_kernel(
    const float* __restrict__ zseq, const _Float16* __restrict__ dW1t,
    const float* __restrict__ db1, const _Float16* __restrict__ dW2t,
    const float* __restrict__ db2, float* __restrict__ out)
{
  __shared__ _Float16 zh[16 * 256];
  __shared__ _Float16 ah[16 * 512];
  const int tid  = threadIdx.x;
  const int lane = tid & 31;
  const int wv   = tid >> 5;
  const size_t row0 = (size_t)blockIdx.x * 16;

  {
    int row = tid >> 4, c0 = (tid & 15) * 16;
    const float* src = zseq + (row0 + row) * 256 + c0;
    for (int i = 0; i < 16; ++i) zh[row * 256 + c0 + i] = (_Float16)src[i];
  }
  __syncthreads();

  v16h afr[8];
  #pragma unroll
  for (int kk = 0; kk < 8; ++kk) afr[kk] = load_afrag_lds(zh, 256, kk * 32, lane);
  #pragma unroll
  for (int t = 0; t < 4; ++t) {
    int n0 = (wv * 4 + t) * 16;          // N = 512
    v8f acc = {};
    #pragma unroll
    for (int kk = 0; kk < 8; ++kk) {
      v16h bf = load_bfrag_g(dW1t, n0, kk * 32, 256, lane);
      acc = wmma_f32(afr[kk], bf, acc);
    }
    int n = n0 + (lane & 15);
    float bb = db1[n];
    #pragma unroll
    for (int r = 0; r < 8; ++r) {
      int M = r + (lane >> 4) * 8;
      ah[M * 512 + n] = (_Float16)tanhf(acc[r] + bb);
    }
  }
  __syncthreads();

  if (wv < 2) {
    int n0 = wv * 16;                    // N = 32
    v8f acc = {};
    #pragma unroll
    for (int kk = 0; kk < 16; ++kk) {    // K = 512
      v16h a2 = load_afrag_lds(ah, 512, kk * 32, lane);
      v16h bf = load_bfrag_g(dW2t, n0, kk * 32, 512, lane);
      acc = wmma_f32(a2, bf, acc);
    }
    int n = n0 + (lane & 15);
    float bb = db2[n];
    #pragma unroll
    for (int r = 0; r < 8; ++r) {
      int M = r + (lane >> 4) * 8;
      out[(row0 + M) * 32 + n] = acc[r] + bb;
    }
  }
}

// ---------- launch ----------

extern "C" void kernel_launch(void* const* d_in, const int* in_sizes, int n_in,
                              void* d_out, int out_size, void* d_ws, size_t ws_size,
                              hipStream_t stream) {
  // input order: ts, us, h0, W1, b1, W2, b2, dW1, db1, dW2, db2
  const float* us  = (const float*)d_in[1];
  const float* h0  = (const float*)d_in[2];
  const float* W1  = (const float*)d_in[3];
  const float* b1  = (const float*)d_in[4];
  const float* W2  = (const float*)d_in[5];
  const float* b2  = (const float*)d_in[6];
  const float* dW1 = (const float*)d_in[7];
  const float* db1 = (const float*)d_in[8];
  const float* dW2 = (const float*)d_in[9];
  const float* db2 = (const float*)d_in[10];

  char* ws = (char*)d_ws;
  // offsets (all 256-byte aligned)
  _Float16* W1t  = (_Float16*)(ws + 0);                  // 256*256*2   = 128 KB
  _Float16* W2t  = (_Float16*)(ws + 131072);             // 4096*256*2  = 2 MB
  _Float16* dW1t = (_Float16*)(ws + 2228224);            // 512*256*2   = 256 KB
  _Float16* dW2t = (_Float16*)(ws + 2490368);            // 32*512*2    = 32 KB
  float*    Dm   = (float*)   (ws + 2523136);            // 255*256*16*4 ~ 4 MB
  float*    zseq = (float*)   (ws + 6701056);            // 256*256*256*4 = 64 MB

  // prep: transpose+cast weights to K-contiguous f16
  transpose_cast_kernel<<<(256 * 256 + 255) / 256, 256, 0, stream>>>(W1, W1t, 256, 256);
  transpose_cast_kernel<<<(256 * 4096 + 255) / 256, 256, 0, stream>>>(W2, W2t, 256, 4096);
  transpose_cast_kernel<<<(256 * 512 + 255) / 256, 256, 0, stream>>>(dW1, dW1t, 256, 512);
  transpose_cast_kernel<<<(512 * 32 + 255) / 256, 256, 0, stream>>>(dW2, dW2t, 512, 32);
  make_dnode_kernel<<<(255 * 256 * 16 + 255) / 256, 256, 0, stream>>>(us, Dm);

  // sequential scan: 16 independent batch blocks, W1 LDS-resident, 16 waves/block
  scan_kernel<<<16, 512, SCAN_SMEM, stream>>>(h0, W1t, b1, W2t, b2, Dm, zseq);

  // readout over all 65536 (l, b) rows
  readout_kernel<<<4096, 256, 0, stream>>>(zseq, dW1t, db1, dW2t, db2, (float*)d_out);
}
